// Encoder_47167330845225
// MI455X (gfx1250) — matrix operand order repaired and checked by reading, not compiled, not run on hardware
//
#include <hip/hip_runtime.h>
#include <hip/hip_bf16.h>
#include <stdint.h>
#include <stddef.h>

// ---------------------------------------------------------------------------
// CDNA5 (gfx1250, wave32) LSTM encoder.
// Core math runs through v_wmma_f32_16x16x32_bf16 (fp32 accumulate).
// ---------------------------------------------------------------------------

typedef __attribute__((ext_vector_type(16))) __bf16 v16bf;
typedef __attribute__((ext_vector_type(8)))  __bf16 bf16x8;
typedef __attribute__((ext_vector_type(8)))  float  v8f;

#define H_   512
#define E_   256
#define XS   264   // sX row stride in bf16 elements (256 + 8 pad, keeps 16B align)
#define HS   520   // sH row stride in bf16 elements (512 + 8 pad)

__device__ __forceinline__ v8f zero_v8f() {
    v8f z;
#pragma unroll
    for (int i = 0; i < 8; ++i) z[i] = 0.0f;
    return z;
}

// A-matrix fragment (16x32 bf16, M x K) in this file's contiguous k order (same as B):
// lanes 0-15: row M=lane, K = k0+0..15 (contiguous, matches B fragment)
// lanes 16-31: row M=lane-16, K = k0+16..31
__device__ __forceinline__ v16bf wmma_load_a(const __bf16* row, int k0, int lhi) {
    bf16x8 lo = *reinterpret_cast<const bf16x8*>(row + k0 + lhi * 16);
    bf16x8 hi = *reinterpret_cast<const bf16x8*>(row + k0 + 8 + lhi * 16);
    v16bf r;
#pragma unroll
    for (int i = 0; i < 8; ++i) { r[i] = lo[i]; r[i + 8] = hi[i]; }
    return r;
}

__device__ __forceinline__ v8f wmma_bf16(v16bf a, v16bf b, v8f c) {
    // (neg_a, A, neg_b, B, c_mod, C, reuse_a, reuse_b)
    return __builtin_amdgcn_wmma_f32_16x16x32_bf16(false, a, false, b, (short)0, c,
                                                   false, false);
}

__device__ __forceinline__ float sigm(float x) { return 1.0f / (1.0f + __expf(-x)); }

// ---------------------------------------------------------------------------
// Persistent recurrent LSTM kernel: block = 16 samples, 8 waves.
// Wave w owns hidden units [w*64, w*64+64) and computes i,f,g,o for them,
// fusing the gate nonlinearity in registers (c kept in WMMA D-layout).
// h is double-buffered in LDS as bf16 for the next step's Whh GEMM.
// Weights are re-streamed from L2 every step (9 MB bf16 total fits easily in
// the 192 MB L2); a memory clobber at loop top prevents LICM from hoisting
// the weight loads (which previously caused massive scratch spills).
// ---------------------------------------------------------------------------
__global__ __launch_bounds__(256) void lstm_wmma_kernel(
    const int*   __restrict__ ids,   // [N][T]
    int T,
    const float* __restrict__ emb,   // [V][E]
    const __bf16* __restrict__ Wih,  // [2048][256] bf16 row-major (gates x E)
    const __bf16* __restrict__ Whh,  // [2048][512] bf16 row-major (gates x H)
    const float* __restrict__ bias,  // [2048]
    float* __restrict__ hout,        // [N][512]
    float* __restrict__ cout_)       // [N][512]
{
    __shared__ __align__(16) __bf16 sX[16 * XS];
    __shared__ __align__(16) __bf16 sH[2][16 * HS];

    const int base  = blockIdx.x * 16;
    const int tid   = threadIdx.x;
    const int wave  = tid >> 5;
    const int lane  = tid & 31;
    const int lr    = lane & 15;
    const int lhi   = lane >> 4;
    const int jbaseW = wave * 64;

    // h0 = c0 = 0
    for (int i = tid; i < 16 * HS; i += 256) {
        sH[0][i] = (__bf16)0.0f;
        sH[1][i] = (__bf16)0.0f;
    }

    // Per-n-tile biases are time-invariant: hoist into registers.
    float bI[4], bF[4], bG[4], bO[4];
#pragma unroll
    for (int nt = 0; nt < 4; ++nt) {
        int j = jbaseW + nt * 16 + lr;
        bI[nt] = bias[j];
        bF[nt] = bias[512 + j];
        bG[nt] = bias[1024 + j];
        bO[nt] = bias[1536 + j];
    }

    v8f cst[4];
#pragma unroll
    for (int nt = 0; nt < 4; ++nt) cst[nt] = zero_v8f();

    int cur = 0;
    for (int t = 0; t < T; ++t) {
        // Forbid hoisting loop-"invariant" global loads (the weight stream)
        // out of the recurrence; without this the compiler spills ~12KB/lane
        // of weight fragments to scratch.
        asm volatile("" ::: "memory");

        __syncthreads();  // prior-step reads of sX done; sH[nxt] writes visible
        // Stage x_t: gather embedding rows, convert f32 -> bf16 into LDS.
        {
            int r  = tid >> 4;           // sample row 0..15
            int cg = (tid & 15) * 16;    // 16-wide column chunk
            int id = ids[(base + r) * T + t];
            const float* src = emb + (size_t)id * E_ + cg;
            __bf16* dst = &sX[r * XS + cg];
#pragma unroll
            for (int u = 0; u < 16; ++u) dst[u] = (__bf16)src[u];
        }
        __syncthreads();

        const __bf16* hcur = sH[cur];
        __bf16*       hnxt = sH[cur ^ 1];
        const __bf16* aXrow = &sX[lr * XS];
        const __bf16* aHrow = &hcur[lr * HS];

#pragma unroll
        for (int nt = 0; nt < 4; ++nt) {
            const int jtb = jbaseW + nt * 16;
            v8f ai = zero_v8f(), af = zero_v8f(), ag = zero_v8f(), ao = zero_v8f();

            // z += x_t @ Wih^T  (K = 256).  B-frag: lane n reads 32B of W row n.
            const __bf16* wi = Wih + (size_t)(jtb + lr) * E_ + lhi * 16;
#pragma unroll 2
            for (int k0 = 0; k0 < E_; k0 += 32) {
                v16bf A  = wmma_load_a(aXrow, k0, lhi);
                v16bf B0 = *reinterpret_cast<const v16bf*>(wi + k0);
                v16bf B1 = *reinterpret_cast<const v16bf*>(wi + 512 * E_ + k0);
                v16bf B2 = *reinterpret_cast<const v16bf*>(wi + 1024 * E_ + k0);
                v16bf B3 = *reinterpret_cast<const v16bf*>(wi + 1536 * E_ + k0);
                ai = wmma_bf16(A, B0, ai);
                af = wmma_bf16(A, B1, af);
                ag = wmma_bf16(A, B2, ag);
                ao = wmma_bf16(A, B3, ao);
            }
            // z += h_{t-1} @ Whh^T  (K = 512)
            const __bf16* wh = Whh + (size_t)(jtb + lr) * H_ + lhi * 16;
#pragma unroll 2
            for (int k0 = 0; k0 < H_; k0 += 32) {
                v16bf A  = wmma_load_a(aHrow, k0, lhi);
                v16bf B0 = *reinterpret_cast<const v16bf*>(wh + k0);
                v16bf B1 = *reinterpret_cast<const v16bf*>(wh + 512 * H_ + k0);
                v16bf B2 = *reinterpret_cast<const v16bf*>(wh + 1024 * H_ + k0);
                v16bf B3 = *reinterpret_cast<const v16bf*>(wh + 1536 * H_ + k0);
                ai = wmma_bf16(A, B0, ai);
                af = wmma_bf16(A, B1, af);
                ag = wmma_bf16(A, B2, ag);
                ao = wmma_bf16(A, B3, ao);
            }

            // Gate fusion in D-layout registers:
            // value (m,n): lane = n + 16*(m>=8), vgpr slot r = m%8.
            const int jcol = jtb + lr;
#pragma unroll
            for (int r = 0; r < 8; ++r) {
                float ii = sigm(ai[r] + bI[nt]);
                float ff = sigm(af[r] + bF[nt]);
                float gg = tanhf(ag[r] + bG[nt]);
                float oo = sigm(ao[r] + bO[nt]);
                float cn = ff * cst[nt][r] + ii * gg;
                cst[nt][r] = cn;
                float hh = oo * tanhf(cn);
                int m = r + lhi * 8;
                hnxt[m * HS + jcol] = (__bf16)hh;
                if (t == T - 1) {
                    hout[(size_t)(base + m) * H_ + jcol]  = hh;
                    cout_[(size_t)(base + m) * H_ + jcol] = cn;
                }
            }
        }
        cur ^= 1;
    }
}

// ---------------------------------------------------------------------------
// Weight precision conversion (once per launch): f32 -> bf16.
// ---------------------------------------------------------------------------
__global__ void f32_to_bf16_kernel(const float* __restrict__ src,
                                   __bf16* __restrict__ dst, int n) {
    int i = blockIdx.x * blockDim.x + threadIdx.x;
    if (i < n) dst[i] = (__bf16)src[i];
}

// ---------------------------------------------------------------------------
// Per-commit scalar merge: hm[i] = [h_sc|h_cm] . Wmh + bmh (and c analog).
// ---------------------------------------------------------------------------
__global__ __launch_bounds__(128) void merge_kernel(
    const float* __restrict__ h_sc, const float* __restrict__ h_cm,
    const float* __restrict__ c_sc, const float* __restrict__ c_cm,
    const float* __restrict__ Wmh, const float* __restrict__ bmh,
    const float* __restrict__ Wmc, const float* __restrict__ bmc,
    float* __restrict__ hm, float* __restrict__ cmm)
{
    __shared__ float rh[128], rc[128];
    const int i = blockIdx.x;  // 0..639
    float ah = 0.f, ac = 0.f;
    for (int k = threadIdx.x; k < H_; k += 128) {
        ah += h_sc[(size_t)i * H_ + k] * Wmh[k] + h_cm[(size_t)i * H_ + k] * Wmh[H_ + k];
        ac += c_sc[(size_t)i * H_ + k] * Wmc[k] + c_cm[(size_t)i * H_ + k] * Wmc[H_ + k];
    }
    rh[threadIdx.x] = ah; rc[threadIdx.x] = ac;
    __syncthreads();
    for (int s = 64; s > 0; s >>= 1) {
        if (threadIdx.x < s) {
            rh[threadIdx.x] += rh[threadIdx.x + s];
            rc[threadIdx.x] += rc[threadIdx.x + s];
        }
        __syncthreads();
    }
    if (threadIdx.x == 0) {
        hm[i]  = rh[0] + bmh[0];
        cmm[i] = rc[0] + bmc[0];
    }
}

// ---------------------------------------------------------------------------
// Final projection: out_h[b][j] = [hm(10) | h_is(512)] . Wfh[j] + bfh[j]
//                   out_c[b][j] = [cm(10) | c_is(512)] . Wfc[j] + bfc[j]
// d_out = [h (64*512) | c (64*512)]
// ---------------------------------------------------------------------------
__global__ __launch_bounds__(256) void final_kernel(
    const float* __restrict__ hm, const float* __restrict__ cmm,
    const float* __restrict__ h_is, const float* __restrict__ c_is,
    const float* __restrict__ Wfh, const float* __restrict__ bfh,
    const float* __restrict__ Wfc, const float* __restrict__ bfc,
    float* __restrict__ out)
{
    __shared__ float sh[H_], sc[H_], shm[10], scm[10];
    const int b = blockIdx.x;  // 0..63
    for (int k = threadIdx.x; k < H_; k += 256) {
        sh[k] = h_is[(size_t)b * H_ + k];
        sc[k] = c_is[(size_t)b * H_ + k];
    }
    if (threadIdx.x < 10) {
        shm[threadIdx.x] = hm[b * 10 + threadIdx.x];
        scm[threadIdx.x] = cmm[b * 10 + threadIdx.x];
    }
    __syncthreads();
    for (int j = threadIdx.x; j < H_; j += 256) {
        float ah = bfh[j], ac = bfc[j];
        const float* wh = Wfh + (size_t)j * 522;
        const float* wc = Wfc + (size_t)j * 522;
#pragma unroll
        for (int k = 0; k < 10; ++k) { ah += shm[k] * wh[k]; ac += scm[k] * wc[k]; }
        for (int k = 0; k < H_; ++k) { ah += sh[k] * wh[10 + k]; ac += sc[k] * wc[10 + k]; }
        out[(size_t)b * H_ + j] = ah;
        out[(size_t)(64 * H_) + (size_t)b * H_ + j] = ac;
    }
}

// ---------------------------------------------------------------------------
extern "C" void kernel_launch(void* const* d_in, const int* in_sizes, int n_in,
                              void* d_out, int out_size, void* d_ws, size_t ws_size,
                              hipStream_t stream) {
    (void)in_sizes; (void)n_in; (void)out_size; (void)ws_size;

    const int*   comments = (const int*)d_in[0];    // [64,10,128]
    const int*   cm_ids   = (const int*)d_in[1];    // [64,10,64]
    const int*   issue    = (const int*)d_in[2];    // [64,32]
    const float* emb_sc   = (const float*)d_in[3];
    const float* emb_cm   = (const float*)d_in[4];
    const float* emb_is   = (const float*)d_in[5];
    const float* Wih_sc   = (const float*)d_in[6];
    const float* Whh_sc   = (const float*)d_in[7];
    const float* b_sc     = (const float*)d_in[8];
    const float* Wih_cm   = (const float*)d_in[9];
    const float* Whh_cm   = (const float*)d_in[10];
    const float* b_cm     = (const float*)d_in[11];
    const float* Wih_is   = (const float*)d_in[12];
    const float* Whh_is   = (const float*)d_in[13];
    const float* b_is     = (const float*)d_in[14];
    const float* Wmh      = (const float*)d_in[15];
    const float* bmh      = (const float*)d_in[16];
    const float* Wmc      = (const float*)d_in[17];
    const float* bmc      = (const float*)d_in[18];
    const float* Wfh      = (const float*)d_in[19];
    const float* bfh      = (const float*)d_in[20];
    const float* Wfc      = (const float*)d_in[21];
    const float* bfc      = (const float*)d_in[22];

    // Workspace carve-out (graph-capture safe: arithmetic only).
    char* ws = (char*)d_ws;
    size_t off = 0;
    auto carve = [&](size_t bytes) -> char* {
        char* p = ws + off;
        off = (off + bytes + 255) & ~(size_t)255;
        return p;
    };
    __bf16* wih_sc_b = (__bf16*)carve((size_t)2048 * 256 * 2);
    __bf16* whh_sc_b = (__bf16*)carve((size_t)2048 * 512 * 2);
    __bf16* wih_cm_b = (__bf16*)carve((size_t)2048 * 256 * 2);
    __bf16* whh_cm_b = (__bf16*)carve((size_t)2048 * 512 * 2);
    __bf16* wih_is_b = (__bf16*)carve((size_t)2048 * 256 * 2);
    __bf16* whh_is_b = (__bf16*)carve((size_t)2048 * 512 * 2);
    float* h_sc = (float*)carve((size_t)640 * 512 * 4);
    float* c_sc = (float*)carve((size_t)640 * 512 * 4);
    float* h_cm = (float*)carve((size_t)640 * 512 * 4);
    float* c_cm = (float*)carve((size_t)640 * 512 * 4);
    float* h_is = (float*)carve((size_t)64 * 512 * 4);
    float* c_is = (float*)carve((size_t)64 * 512 * 4);
    float* hm   = (float*)carve((size_t)640 * 4);
    float* cmm  = (float*)carve((size_t)640 * 4);

    // 1) Weights -> bf16 (stay resident in L2 thereafter; 9 MB total).
    const int nE = 2048 * 256, nH = 2048 * 512;
    f32_to_bf16_kernel<<<(nE + 255) / 256, 256, 0, stream>>>(Wih_sc, wih_sc_b, nE);
    f32_to_bf16_kernel<<<(nH + 255) / 256, 256, 0, stream>>>(Whh_sc, whh_sc_b, nH);
    f32_to_bf16_kernel<<<(nE + 255) / 256, 256, 0, stream>>>(Wih_cm, wih_cm_b, nE);
    f32_to_bf16_kernel<<<(nH + 255) / 256, 256, 0, stream>>>(Whh_cm, whh_cm_b, nH);
    f32_to_bf16_kernel<<<(nE + 255) / 256, 256, 0, stream>>>(Wih_is, wih_is_b, nE);
    f32_to_bf16_kernel<<<(nH + 255) / 256, 256, 0, stream>>>(Whh_is, whh_is_b, nH);

    // 2) Recurrent WMMA LSTMs (640/16 = 40 blocks, 64/16 = 4 blocks).
    lstm_wmma_kernel<<<40, 256, 0, stream>>>(comments, 128, emb_sc,
                                             wih_sc_b, whh_sc_b, b_sc, h_sc, c_sc);
    lstm_wmma_kernel<<<40, 256, 0, stream>>>(cm_ids, 64, emb_cm,
                                             wih_cm_b, whh_cm_b, b_cm, h_cm, c_cm);
    lstm_wmma_kernel<<<4, 256, 0, stream>>>(issue, 32, emb_is,
                                            wih_is_b, whh_is_b, b_is, h_is, c_is);

    // 3) Per-commit scalar merge, then final projection into d_out = [h | c].
    merge_kernel<<<640, 128, 0, stream>>>(h_sc, h_cm, c_sc, c_cm,
                                          Wmh, bmh, Wmc, bmc, hm, cmm);
    final_kernel<<<64, 256, 0, stream>>>(hm, cmm, h_is, c_is,
                                         Wfh, bfh, Wfc, bfc, (float*)d_out);
}
